// cubs2_76725295775783
// MI455X (gfx1250) — matrix operand order repaired and checked by reading, not compiled, not run on hardware
//
#include <hip/hip_runtime.h>

#define BN 16
#define CC 256
#define NN 4096   // H*W = 64*64

typedef __attribute__((ext_vector_type(16))) _Float16 v16h;
typedef __attribute__((ext_vector_type(8)))  float    v8f;

#define LOG2E 1.44269504088896340736f

static __device__ __forceinline__ unsigned pk2h(float lo, float hi) {
    // __builtin_amdgcn_cvt_pkrtz returns __fp16 ext_vector(2) -> v_cvt_pkrtz_f16_f32
    typedef __fp16 fp16x2 __attribute__((ext_vector_type(2)));
    fp16x2 p = __builtin_amdgcn_cvt_pkrtz(lo, hi);
    return __builtin_bit_cast(unsigned, p);
}

// Raw hardware base-2 exponent (v_exp_f32), no OCML range guard. Arguments here
// are always <= 0 (max-subtracted), where hardware flushes underflow to 0 --
// exactly the softmax semantics we need.
static __device__ __forceinline__ float hexp2(float x) {
    return __builtin_amdgcn_exp2f(x);
}

// ---------------- Kernel 1: x0/x1/x2 projections (one row per wave32) --------
__global__ __launch_bounds__(256) void proj_kernel(
    const float* __restrict__ x,
    const float* __restrict__ w1, const float* __restrict__ b1,
    const float* __restrict__ w2, const float* __restrict__ b2,
    const float* __restrict__ w3, const float* __restrict__ b3,
    float* __restrict__ x0g, float* __restrict__ x1g, float* __restrict__ x2g)
{
    __shared__ float4 lw1[CC/4], lw2[CC/4], lw3[CC/4];
    const int tid = threadIdx.x;
    if (tid < CC/4) {
        lw1[tid] = ((const float4*)w1)[tid];
        lw2[tid] = ((const float4*)w2)[tid];
        lw3[tid] = ((const float4*)w3)[tid];
    }
    __syncthreads();

    const int wave = tid >> 5;
    const int lane = tid & 31;
    const long row = (long)blockIdx.x * 8 + wave;     // 8 waves per block

    const float4* rp = (const float4*)(x + row * CC);
    float4 xa = rp[lane];
    float4 xb = rp[lane + 32];

    float4 wa, wb;
    wa = lw1[lane]; wb = lw1[lane + 32];
    float s0 = xa.x*wa.x + xa.y*wa.y + xa.z*wa.z + xa.w*wa.w
             + xb.x*wb.x + xb.y*wb.y + xb.z*wb.z + xb.w*wb.w;
    wa = lw2[lane]; wb = lw2[lane + 32];
    float s1 = xa.x*wa.x + xa.y*wa.y + xa.z*wa.z + xa.w*wa.w
             + xb.x*wb.x + xb.y*wb.y + xb.z*wb.z + xb.w*wb.w;
    wa = lw3[lane]; wb = lw3[lane + 32];
    float s2 = xa.x*wa.x + xa.y*wa.y + xa.z*wa.z + xa.w*wa.w
             + xb.x*wb.x + xb.y*wb.y + xb.z*wb.z + xb.w*wb.w;

    #pragma unroll
    for (int off = 16; off > 0; off >>= 1) {
        s0 += __shfl_xor(s0, off, 32);
        s1 += __shfl_xor(s1, off, 32);
        s2 += __shfl_xor(s2, off, 32);
    }
    if (lane == 0) {
        x0g[row] = s0 + b1[0];
        x1g[row] = s1 + b2[0];
        x2g[row] = s2 + b3[0];
    }
}

// ---------------- Kernel 2: per-batch max/min of x0 --------------------------
__global__ __launch_bounds__(256) void minmax_kernel(
    const float* __restrict__ x0g, float* __restrict__ bmax, float* __restrict__ bmin)
{
    __shared__ float smax[256], smin[256];
    const int b = blockIdx.x, tid = threadIdx.x;
    float mx = -3.0e38f, mn = 3.0e38f;
    for (int i = tid; i < NN; i += 256) {
        float v = x0g[b * NN + i];
        mx = fmaxf(mx, v); mn = fminf(mn, v);
    }
    smax[tid] = mx; smin[tid] = mn; __syncthreads();
    for (int s = 128; s > 0; s >>= 1) {
        if (tid < s) {
            smax[tid] = fmaxf(smax[tid], smax[tid + s]);
            smin[tid] = fminf(smin[tid], smin[tid + s]);
        }
        __syncthreads();
    }
    if (tid == 0) { bmax[b] = smax[0]; bmin[b] = smin[0]; }
}

// ---------------- Kernel 3: softmax-weighted feats via WMMA ------------------
// Each wave32 owns 16 output columns m. Loop n in chunks of 32:
//   B operand (32x16 f16) = exp2(x0[n]*t'[m] - mx'[m]) tile  (t' prescaled by log2e)
//   A operand (16x32 f16) : row0 = x2 chunk, row1 = ones, rows 2..15 = 0
//   C (16x16 f32) accumulates: row0 = numerator, row1 = denominator.
// LDS reads are unconditional float4; A row-select is a dword cndmask against a
// hoisted per-lane constant, so EXEC stays all-ones through the whole loop.
__global__ __launch_bounds__(256) void feats_kernel(
    const float* __restrict__ x0g, const float* __restrict__ x1g,
    const float* __restrict__ x2g, const float* __restrict__ bmax,
    const float* __restrict__ bmin, float* __restrict__ gate)
{
    __shared__ float lx0[NN];
    __shared__ float lx2[NN];
    const int b   = blockIdx.y;
    const int tid = threadIdx.x;

    for (int i = tid; i < NN/4; i += 256) {
        ((float4*)lx0)[i] = ((const float4*)(x0g + (long)b * NN))[i];
        ((float4*)lx2)[i] = ((const float4*)(x2g + (long)b * NN))[i];
    }
    __syncthreads();

    const int wave = tid >> 5;
    const int lane = tid & 31;
    const int m    = lane & 15;    // column within tile (B/C layout: col = lane%16)
    const int g    = lane >> 4;    // lane-half selects K range
    const int col  = blockIdx.x * 128 + wave * 16 + m;

    const float t   = x1g[(long)b * NN + col];
    const float mxv = (t >= 0.0f) ? t * bmax[b] : t * bmin[b];
    const float tl  = t * LOG2E;      // fold ln->log2 into the fma
    const float mxl = mxv * LOG2E;

    // Hoisted per-lane A-operand constant: row1 = ones (0x3C00 pairs), rows>=2 = 0.
    const unsigned aconst = (m == 1) ? 0x3C003C00u : 0u;
    const bool     isrow0 = (m == 0);

    union { v16h h; unsigned u[8]; } A, Bm;
    v8f acc = {};
    for (int base = 0; base < NN; base += 32) {
        // B operand source: x0[base + g*16 + h], h = 0..15 (contiguous 16)
        const float4* bp = (const float4*)(lx0 + base + g * 16);
        float4 q0 = bp[0], q1 = bp[1], q2 = bp[2], q3 = bp[3];
        // A operand source: x2 at K = 16g + h, h = 0..15 (contiguous 16)
        const float4* apA = (const float4*)(lx2 + base + 16 * g);
        const float4* apB = (const float4*)(lx2 + base + 16 * g + 8);
        float4 r0 = apA[0], r1 = apA[1], r2 = apB[0], r3 = apB[1];

        float xb[16] = {q0.x, q0.y, q0.z, q0.w, q1.x, q1.y, q1.z, q1.w,
                        q2.x, q2.y, q2.z, q2.w, q3.x, q3.y, q3.z, q3.w};
        float xa[16] = {r0.x, r0.y, r0.z, r0.w, r1.x, r1.y, r1.z, r1.w,
                        r2.x, r2.y, r2.z, r2.w, r3.x, r3.y, r3.z, r3.w};

        #pragma unroll
        for (int d = 0; d < 8; ++d) {
            // exp tile -> B operand (v_fma + v_exp_f32 + v_cvt_pk_rtz)
            float e0 = hexp2(xb[2*d]     * tl - mxl);
            float e1 = hexp2(xb[2*d + 1] * tl - mxl);
            Bm.u[d] = pk2h(e0, e1);
            // x2 row -> A operand (v_cvt_pk_rtz + v_cndmask)
            unsigned w = pk2h(xa[2*d], xa[2*d + 1]);
            A.u[d] = isrow0 ? w : aconst;
        }
        acc = __builtin_amdgcn_wmma_f32_16x16x32_f16(
            /*neg_a=*/false, A.h, /*neg_b=*/false, Bm.h,
            /*c_mod=*/(short)0, acc, /*reuse_a=*/false, /*reuse_b=*/false);
    }

    // C/D layout: VGPR r, lanes 0-15 hold row M=r  -> acc[0]=num, acc[1]=den
    if (lane < 16) {
        float f = acc[0] * __builtin_amdgcn_rcpf(acc[1]);          // num/den
        float s = __builtin_amdgcn_rcpf(1.0f + hexp2(-f * LOG2E)); // sigmoid
        gate[(long)b * NN + col] = s;
    }
}

// ---------------- Kernel 4: out = sigmoid(feats) * x (float4 streaming) ------
__global__ __launch_bounds__(256) void gate_kernel(
    const float* __restrict__ x, const float* __restrict__ gate,
    float* __restrict__ out)
{
    const long i   = (long)blockIdx.x * 256 + threadIdx.x;  // float4 index
    const long pix = i / (CC / 4);
    const float gv = gate[pix];
    float4 xv = ((const float4*)x)[i];
    float4 o;
    o.x = xv.x * gv; o.y = xv.y * gv; o.z = xv.z * gv; o.w = xv.w * gv;
    ((float4*)out)[i] = o;
}

extern "C" void kernel_launch(void* const* d_in, const int* in_sizes, int n_in,
                              void* d_out, int out_size, void* d_ws, size_t ws_size,
                              hipStream_t stream) {
    const float* x  = (const float*)d_in[0];
    const float* w1 = (const float*)d_in[1];
    const float* b1 = (const float*)d_in[2];
    const float* w2 = (const float*)d_in[3];
    const float* b2 = (const float*)d_in[4];
    const float* w3 = (const float*)d_in[5];
    const float* b3 = (const float*)d_in[6];

    float* ws   = (float*)d_ws;
    float* x0g  = ws;                       // B*N
    float* x1g  = ws + (long)BN * NN;       // B*N
    float* x2g  = ws + 2L * BN * NN;        // B*N
    float* gate = ws + 3L * BN * NN;        // B*N
    float* bmax = ws + 4L * BN * NN;        // B
    float* bmin = bmax + BN;                // B

    // Phase A: projections (8 rows per 256-thread block)
    proj_kernel<<<BN * NN / 8, 256, 0, stream>>>(x, w1, b1, w2, b2, w3, b3,
                                                 x0g, x1g, x2g);
    // Phase A2: per-batch max/min of x0 for stable softmax
    minmax_kernel<<<BN, 256, 0, stream>>>(x0g, bmax, bmin);
    // Phase B: 128 columns per block, B batches
    feats_kernel<<<dim3(NN / 128, BN), 256, 0, stream>>>(x0g, x1g, x2g,
                                                         bmax, bmin, gate);
    // Phase C: gated output
    gate_kernel<<<(int)((long)BN * NN * CC / 4 / 256), 256, 0, stream>>>(
        x, gate, (float*)d_out);
}
